// MambaBlock_16544214024238
// MI455X (gfx1250) — compile-verified, matrix operand order checked by reading, not run on hardware
//
#include <hip/hip_runtime.h>
#include <math.h>

typedef _Float16 v16h __attribute__((ext_vector_type(16)));
typedef _Float16 v8h  __attribute__((ext_vector_type(8)));
typedef float    v8f  __attribute__((ext_vector_type(8)));

#define D_MODEL 1024
#define D_INNER 2048
#define D_STATE 16
#define DT_RANK 64
#define D_CONV  4
#define BATCH   2
#define SEQ     1024
#define ROWS    (BATCH * SEQ)   // 2048 tokens

// ---------------------------------------------------------------------------
// Generic f16 WMMA GEMM: out[M,N] = A[M,K] (row-major f16) x W[N,K]^T (row-major f16)
// Register-blocked: each wave32 computes a 16(M) x NT*16(N) strip = NT
// accumulator tiles, reusing one A fragment across NT B fragments.
// NT and the epilogue MODE are compile-time so the k-loop is branch-free
// (EXEC all-ones, no phi-copies, compiler can software-pipeline).
// 4 waves per block stack in M -> block tile is 64 x NT*16.
// Column coverage starts at nbase (remainder columns get a second launch).
// Epilogue modes:
//   0: store f16 to outH (ld = N)
//   1: store f32 to outF (ld = N) and, for cols < 64, f16 copy to outH (ld = 64)
//   2: store f32 to outF (ld = N)
//   3: store f32 to outF (ld = N) with residual add from resid (ld = N)
// ---------------------------------------------------------------------------
template <int NT, int MODE>
__global__ __launch_bounds__(128)
void wmma_gemm_f16(const _Float16* __restrict__ A, const _Float16* __restrict__ W,
                   float* __restrict__ outF, _Float16* __restrict__ outH,
                   const float* __restrict__ resid,
                   int M, int N, int K, int nbase)
{
    const int lane = threadIdx.x & 31;
    const int wave = threadIdx.x >> 5;
    const int m0 = (blockIdx.y * 4 + wave) * 16;       // 16-row strip per wave
    const int n0 = nbase + blockIdx.x * (NT * 16);     // NT x 16-col tiles
    if (m0 >= M) return;                               // wave-uniform

    // f16 A/B fragment layout (ISA 7.12.2):
    //   lane%16 selects the row (M for A, N for B);
    //   lanes <16 hold K {0..7, 16..23}, lanes >=16 hold K {8..15, 24..31}
    const int row = lane & 15;
    const int kh  = (lane >> 4) << 3;                  // 0 or 8

    const _Float16* ap = A + (size_t)(m0 + row) * K + kh;
    const _Float16* wp = W + (size_t)(n0 + row) * K + kh;
    const size_t wstep = (size_t)16 * K;               // 16 N-rows per sub-tile

    v8f acc[NT] = {};
    for (int k0 = 0; k0 < K; k0 += 32) {
        __builtin_prefetch(ap + k0 + 128, 0, 1);
        v8h a_lo = *(const v8h*)(ap + k0);
        v8h a_hi = *(const v8h*)(ap + k0 + 16);
        v16h av = __builtin_shufflevector(a_lo, a_hi,
                    0,1,2,3,4,5,6,7,8,9,10,11,12,13,14,15);
        #pragma unroll
        for (int j = 0; j < NT; ++j) {
            v8h b_lo = *(const v8h*)(wp + j * wstep + k0);
            v8h b_hi = *(const v8h*)(wp + j * wstep + k0 + 16);
            v16h bv = __builtin_shufflevector(b_lo, b_hi,
                        0,1,2,3,4,5,6,7,8,9,10,11,12,13,14,15);
            acc[j] = __builtin_amdgcn_wmma_f32_16x16x32_f16(
                         false, av, false, bv, (short)0, acc[j], false, false);
        }
    }

    // C/D layout: element v of lane l -> (M = m0 + 8*(l>=16) + v, N = n0 + l%16)
    const int rbase = m0 + ((lane >> 4) << 3);
    #pragma unroll
    for (int j = 0; j < NT; ++j) {
        const int c = n0 + j * 16 + row;
        #pragma unroll
        for (int v = 0; v < 8; ++v) {
            const size_t idx = (size_t)(rbase + v) * N + c;
            const float x = acc[j][v];
            if (MODE == 0) {
                outH[idx] = (_Float16)x;
            } else if (MODE == 1) {
                outF[idx] = x;
                if (c < 64) outH[(size_t)(rbase + v) * 64 + c] = (_Float16)x;
            } else if (MODE == 2) {
                outF[idx] = x;
            } else {
                outF[idx] = x + resid[idx];
            }
        }
    }
}

// ---------------------------------------------------------------------------
// RMSNorm over D_MODEL, one token per block, output f16 for the WMMA GEMM.
// ---------------------------------------------------------------------------
__global__ __launch_bounds__(256)
void rmsnorm_kernel(const float* __restrict__ u, const float* __restrict__ w_norm,
                    _Float16* __restrict__ out)
{
    const int rowid = blockIdx.x;
    const float* ur = u + (size_t)rowid * D_MODEL;
    float ss = 0.f;
    for (int c = threadIdx.x; c < D_MODEL; c += 256) { float v = ur[c]; ss += v * v; }
    #pragma unroll
    for (int o = 16; o > 0; o >>= 1) ss += __shfl_down(ss, o, 32);  // wave32
    __shared__ float red[8];
    __shared__ float inv_s;
    if ((threadIdx.x & 31) == 0) red[threadIdx.x >> 5] = ss;
    __syncthreads();
    if (threadIdx.x == 0) {
        float t = 0.f;
        #pragma unroll
        for (int w = 0; w < 8; ++w) t += red[w];
        inv_s = rsqrtf(t / (float)D_MODEL + 1e-5f);
    }
    __syncthreads();
    const float inv = inv_s;
    for (int c = threadIdx.x; c < D_MODEL; c += 256)
        out[(size_t)rowid * D_MODEL + c] = (_Float16)(w_norm[c] * ur[c] * inv);
}

// ---------------------------------------------------------------------------
// f32 -> f16 weight cast
// ---------------------------------------------------------------------------
__global__ void cast_f32_f16(const float* __restrict__ in, _Float16* __restrict__ out, int n)
{
    int i = blockIdx.x * blockDim.x + threadIdx.x;
    if (i < n) out[i] = (_Float16)in[i];
}

// ---------------------------------------------------------------------------
// Depthwise causal conv (width 4) + bias + SiLU on the x half of xz.
// ---------------------------------------------------------------------------
__global__ __launch_bounds__(256)
void conv_silu_kernel(const _Float16* __restrict__ xz_h, const float* __restrict__ conv_w,
                      const float* __restrict__ conv_b, _Float16* __restrict__ x_h)
{
    int i = blockIdx.x * blockDim.x + threadIdx.x;
    if (i >= BATCH * SEQ * D_INNER) return;
    const int d = i % D_INNER;
    const int t = (i / D_INNER) % SEQ;
    const int b = i / (D_INNER * SEQ);
    float acc = conv_b[d];
    #pragma unroll
    for (int j = 0; j < D_CONV; ++j) {
        const int tt = t - (D_CONV - 1) + j;
        if (tt >= 0)
            acc += conv_w[d * D_CONV + j] *
                   (float)xz_h[(size_t)(b * SEQ + tt) * (2 * D_INNER) + d];
    }
    const float sv = acc / (1.f + __expf(-acc));   // silu
    x_h[(size_t)(b * SEQ + t) * D_INNER + d] = (_Float16)sv;
}

// ---------------------------------------------------------------------------
// Selective scan: one (batch, channel) per lane; 16-wide state in registers.
// dt/B/C kept in f32 for accuracy; B/C rows (96 floats/token) are shared by
// all 2048 channels of a batch so they live in L2/WGP$.
// ---------------------------------------------------------------------------
__global__ __launch_bounds__(128)
void scan_kernel(const float* __restrict__ dt_lin, const _Float16* __restrict__ x_h,
                 const _Float16* __restrict__ xz_h, const float* __restrict__ xp,
                 const float* __restrict__ A_log, const float* __restrict__ b_dt,
                 const float* __restrict__ D_param, _Float16* __restrict__ y_h)
{
    const int idx = blockIdx.x * blockDim.x + threadIdx.x;
    if (idx >= BATCH * D_INNER) return;
    const int b = idx / D_INNER;
    const int d = idx % D_INNER;

    float Arow[D_STATE];
    #pragma unroll
    for (int n = 0; n < D_STATE; ++n) Arow[n] = -__expf(A_log[d * D_STATE + n]);
    const float bdt = b_dt[d];
    const float Dp  = D_param[d];

    float s[D_STATE];
    #pragma unroll
    for (int n = 0; n < D_STATE; ++n) s[n] = 0.f;

    for (int t = 0; t < SEQ; ++t) {
        const size_t r = (size_t)b * SEQ + t;
        const float dtl = dt_lin[r * D_INNER + d] + bdt;
        const float dt  = (dtl > 20.f) ? dtl : log1pf(__expf(dtl));   // softplus
        const float xv  = (float)x_h[r * D_INNER + d];
        const float zv  = (float)xz_h[r * (2 * D_INNER) + D_INNER + d];
        const float* xpr = xp + r * (DT_RANK + 2 * D_STATE);
        float y = 0.f;
        #pragma unroll
        for (int n = 0; n < D_STATE; ++n) {
            const float dA = __expf(dt * Arow[n]);
            s[n] = dA * s[n] + (dt * xpr[DT_RANK + n]) * xv;          // dBu
            y += s[n] * xpr[DT_RANK + D_STATE + n];                    // y = s·C
        }
        y = (y + Dp * xv) * (zv / (1.f + __expf(-zv)));                // gate with silu(z)
        y_h[r * D_INNER + d] = (_Float16)y;
    }
}

// ---------------------------------------------------------------------------
extern "C" void kernel_launch(void* const* d_in, const int* in_sizes, int n_in,
                              void* d_out, int out_size, void* d_ws, size_t ws_size,
                              hipStream_t stream)
{
    const float* u      = (const float*)d_in[0];
    const float* w_norm = (const float*)d_in[1];
    const float* w_in   = (const float*)d_in[2];
    const float* conv_w = (const float*)d_in[3];
    const float* conv_b = (const float*)d_in[4];
    const float* w_x    = (const float*)d_in[5];
    const float* w_dt   = (const float*)d_in[6];
    const float* b_dt   = (const float*)d_in[7];
    const float* A_log  = (const float*)d_in[8];
    const float* D_par  = (const float*)d_in[9];
    const float* w_out  = (const float*)d_in[10];
    (void)in_sizes; (void)n_in; (void)out_size; (void)ws_size;

    char* p = (char*)d_ws;
    auto take = [&](size_t bytes) -> char* {
        char* q = p; p += (bytes + 255) & ~(size_t)255; return q;
    };
    _Float16* w_in_h  = (_Float16*)take((size_t)2 * D_INNER * D_MODEL * 2);
    _Float16* w_x_h   = (_Float16*)take((size_t)(DT_RANK + 2 * D_STATE) * D_INNER * 2);
    _Float16* w_out_h = (_Float16*)take((size_t)D_MODEL * D_INNER * 2);
    _Float16* w_dt_h  = (_Float16*)take((size_t)D_INNER * DT_RANK * 2);
    _Float16* unorm_h = (_Float16*)take((size_t)ROWS * D_MODEL * 2);
    _Float16* xz_h    = (_Float16*)take((size_t)ROWS * 2 * D_INNER * 2);
    _Float16* x_h     = (_Float16*)take((size_t)ROWS * D_INNER * 2);
    float*    xp_f    = (float*)   take((size_t)ROWS * (DT_RANK + 2 * D_STATE) * 4);
    _Float16* dtr_h   = (_Float16*)take((size_t)ROWS * DT_RANK * 2);
    float*    dtl_f   = (float*)   take((size_t)ROWS * D_INNER * 4);
    _Float16* y_h     = (_Float16*)take((size_t)ROWS * D_INNER * 2);

    // Weight casts f32 -> f16
    {
        int n;
        n = 2 * D_INNER * D_MODEL;
        cast_f32_f16<<<(n + 255) / 256, 256, 0, stream>>>(w_in, w_in_h, n);
        n = (DT_RANK + 2 * D_STATE) * D_INNER;
        cast_f32_f16<<<(n + 255) / 256, 256, 0, stream>>>(w_x, w_x_h, n);
        n = D_MODEL * D_INNER;
        cast_f32_f16<<<(n + 255) / 256, 256, 0, stream>>>(w_out, w_out_h, n);
        n = D_INNER * DT_RANK;
        cast_f32_f16<<<(n + 255) / 256, 256, 0, stream>>>(w_dt, w_dt_h, n);
    }

    // 1) RMSNorm -> f16
    rmsnorm_kernel<<<ROWS, 256, 0, stream>>>(u, w_norm, unorm_h);

    const dim3 blk(128);
    const int gy = ROWS / 64;   // 32 (M = 2048, multiple of 64)

    // 2) xz = u_norm @ w_in^T : [2048 x 4096], K=1024 -> f16
    wmma_gemm_f16<4, 0><<<dim3(4096 / 64, gy), blk, 0, stream>>>(
        unorm_h, w_in_h, nullptr, xz_h, nullptr, ROWS, 2 * D_INNER, D_MODEL, 0);

    // 3) depthwise conv + silu -> x f16
    {
        int n = BATCH * SEQ * D_INNER;
        conv_silu_kernel<<<(n + 255) / 256, 256, 0, stream>>>(xz_h, conv_w, conv_b, x_h);
    }

    // 4) xp = x @ w_x^T : [2048 x 96], K=2048 -> f32 (+ f16 dt_r slice)
    //    cols 0..63 (NT=4) then cols 64..95 (NT=2)
    wmma_gemm_f16<4, 1><<<dim3(1, gy), blk, 0, stream>>>(
        x_h, w_x_h, xp_f, dtr_h, nullptr, ROWS, 96, D_INNER, 0);
    wmma_gemm_f16<2, 1><<<dim3(1, gy), blk, 0, stream>>>(
        x_h, w_x_h, xp_f, dtr_h, nullptr, ROWS, 96, D_INNER, 64);

    // 5) dt_lin = dt_r @ w_dt^T : [2048 x 2048], K=64 -> f32
    wmma_gemm_f16<4, 2><<<dim3(2048 / 64, gy), blk, 0, stream>>>(
        dtr_h, w_dt_h, dtl_f, nullptr, nullptr, ROWS, D_INNER, DT_RANK, 0);

    // 6) selective scan + D skip + silu(z) gate -> y f16
    {
        int n = BATCH * D_INNER;
        scan_kernel<<<(n + 127) / 128, 128, 0, stream>>>(
            dtl_f, x_h, xz_h, xp_f, A_log, b_dt, D_par, y_h);
    }

    // 7) out = y @ w_out^T + u : [2048 x 1024], K=2048 -> f32 (d_out)
    wmma_gemm_f16<4, 3><<<dim3(1024 / 64, gy), blk, 0, stream>>>(
        y_h, w_out_h, (float*)d_out, nullptr, u, ROWS, D_MODEL, D_INNER, 0);
}
